// SentenceEmbedding_37280316129500
// MI455X (gfx1250) — matrix operand order repaired and checked by reading, not compiled, not run on hardware
//
#include <hip/hip_runtime.h>
#include <hip/hip_bf16.h>

typedef __attribute__((ext_vector_type(16))) _Float16 v16h;
typedef __attribute__((ext_vector_type(8)))  _Float16 v8h;
typedef __attribute__((ext_vector_type(8)))  float    v8f;

#define BATCH   256
#define TLEN    128
#define EDIM    256
#define HDIM    256
#define KDIM    512      // E + H
#define GDIM    1024     // 4H
#define VOCAB   32000
#define BBLK    32       // batch rows per block (2 WMMA row-tiles)

// Hardware V_TANH_F32 on gfx1250 (single TRANS op, co-executes with WMMA);
// guarded so the file still compiles if the builtin is absent.
__device__ __forceinline__ float fast_tanhf(float x) {
#if __has_builtin(__builtin_amdgcn_tanhf)
    return __builtin_amdgcn_tanhf(x);
#else
    return tanhf(x);
#endif
}

// Exact identity: sigmoid(x) = 0.5*tanh(x/2) + 0.5  -> 1 TRANS + 2 VALU ops.
__device__ __forceinline__ float sigmoidf_(float x) {
    return fmaf(0.5f, fast_tanhf(0.5f * x), 0.5f);
}

__device__ __forceinline__ v16h mk16(v8h lo, v8h hi) {
    return __builtin_shufflevector(lo, hi,
           0, 1, 2, 3, 4, 5, 6, 7, 8, 9, 10, 11, 12, 13, 14, 15);
}

// ---------------- prep kernel 1: Emb fp32 -> f16 ----------------
__global__ __launch_bounds__(256) void emb_to_h(const float* __restrict__ emb,
                                                _Float16* __restrict__ out) {
    int i = blockIdx.x * 256 + threadIdx.x;   // VOCAB*EDIM = 8,192,000
    out[i] = (_Float16)emb[i];
}

// ---------------- prep kernel 2: build transposed f16 [W;U] slabs ------------
// WT6 layout: slab(l*2+dir) x [n=0..1023][k=0..511], k<256 -> W[k][n], else U[k-256][n]
__global__ __launch_bounds__(256) void build_wt(const float* __restrict__ Wf,
                                                const float* __restrict__ Uf,
                                                const float* __restrict__ Wb,
                                                const float* __restrict__ Ub,
                                                _Float16* __restrict__ WT6) {
    int idx = blockIdx.x * 256 + threadIdx.x;   // 6*1024*512 = 3,145,728
    int slab = idx >> 19;            // 1024*512 = 2^19
    int rem  = idx & ((1 << 19) - 1);
    int n = rem >> 9;
    int k = rem & 511;
    int l = slab >> 1;
    int d = slab & 1;
    const float* W = d ? Wb : Wf;
    const float* U = d ? Ub : Uf;
    float v = (k < EDIM) ? W[(size_t)(l * EDIM + k) * GDIM + n]
                         : U[(size_t)(l * HDIM + (k - EDIM)) * GDIM + n];
    WT6[idx] = (_Float16)v;
}

// ---------------- main kernel: fused BiLSTM recurrence + max pool ------------
// grid: (8 batch-blocks of 32 rows, 6 = layer*2+dir), block: 512 threads = 16 waves.
// Each wave: 2 row-tiles x 4 gate-tiles (one 16-col chunk per gate -> lane-local
// LSTM cell update). Depth-2 software pipeline on k-chunks keeps the next
// chunk's A/B fragments in flight while the current chunk's 8 WMMAs execute.
__global__ __launch_bounds__(512) void bilstm_pool(const int* __restrict__ tokens,
                                                   const _Float16* __restrict__ EmbH,
                                                   const _Float16* __restrict__ WT6,
                                                   const float* __restrict__ bf,
                                                   const float* __restrict__ bb,
                                                   float* __restrict__ out) {
    const int bx   = blockIdx.x;             // batch block 0..7
    const int slab = blockIdx.y;             // l*2 + dir
    const int l    = slab >> 1;
    const int dir  = slab & 1;
    const _Float16* WT   = WT6 + (size_t)slab * GDIM * KDIM;
    const float*    bias = (dir ? bb : bf) + l * GDIM;

    const int tid    = threadIdx.x;
    const int wave   = tid >> 5;             // 0..15 (wave32)
    const int lane   = tid & 31;
    const int laneN  = lane & 15;            // column-within-tile / A row
    const int laneHi = lane >> 4;            // 0 or 1

    // double-buffered A tile: buf[i] = [row 0..31][ x(0..255) | h(256..511) ]
    __shared__ _Float16 Alds[2 * BBLK * KDIM];   // 64 KB

    const int gRow   = tid >> 4;             // gather row 0..31
    const int gChunk = tid & 15;             // gather 16-half chunk

    // ---- prologue: gather x_0 into buf0, zero buf0 h-region ----
    {
        int t0  = dir ? (TLEN - 1) : 0;
        int tok = tokens[(bx * BBLK + gRow) * TLEN + t0];
        unsigned xoff = (unsigned)tok * EDIM + gChunk * 16;   // 32-bit offset
        const v8h* src = (const v8h*)(EmbH + xoff);
        v8h* dst = (v8h*)(Alds + gRow * KDIM + gChunk * 16);
        dst[0] = src[0];
        dst[1] = src[1];
        #pragma unroll
        for (int it = 0; it < 16; ++it) {
            int j = it * 512 + tid;          // 32*256 = 8192 h elements
            int r = j >> 8, c = j & 255;
            Alds[r * KDIM + EDIM + c] = (_Float16)0.0f;
        }
    }

    // per-wave gate tiles: tile g -> columns [g*256 + wave*16, +16)
    // 32-bit half-offsets into WT (keeps B loads in SADDR + voffset form)
    int bOffH[4];
    float biasv[4];
    #pragma unroll
    for (int g = 0; g < 4; ++g) {
        int n0 = g * HDIM + wave * 16;
        bOffH[g] = (n0 + laneN) * KDIM + laneHi * 8;
        biasv[g] = bias[n0 + laneN];
    }

    float c_state[16];                       // [rowtile(2)][row(8)]
    float hmax[16];
    #pragma unroll
    for (int i = 0; i < 16; ++i) { c_state[i] = 0.0f; hmax[i] = -INFINITY; }

    const int kbA   = laneHi * 8;            // A k sub-offset per lane group
    const int aOff0 = laneN * KDIM + kbA;            // row-tile 0 (rows 0..15)
    const int aOff1 = (16 + laneN) * KDIM + kbA;     // row-tile 1 (rows 16..31)

    __syncthreads();                         // buf0 ready

    for (int step = 0; step < TLEN; ++step) {
        const int cur = step & 1;
        const int nxt = cur ^ 1;
        _Float16* Acur = Alds + cur * (BBLK * KDIM);
        _Float16* Anxt = Alds + nxt * (BBLK * KDIM);

        // issue next step's x-gather into registers (latency hidden by GEMM)
        v8h xa, xb;
        const bool haveNext = (step + 1) < TLEN;
        if (haveNext) {
            int t2  = dir ? (TLEN - 2 - step) : (step + 1);
            int tok = tokens[(bx * BBLK + gRow) * TLEN + t2];
            unsigned xoff = (unsigned)tok * EDIM + gChunk * 16;
            const v8h* src = (const v8h*)(EmbH + xoff);
            xa = src[0];
            xb = src[1];
        }

        // z = [x_t | h] * [W;U] + b  via 16x16x32 f16 WMMA, K = 512.
        // k-chunk order rotates with step (defeats LICM / scratch spilling);
        // explicit depth-2 pipeline keeps next chunk's loads in flight.
        v8f acc[8];                          // [rowtile(2)][gate(4)]
        #pragma unroll
        for (int rt = 0; rt < 2; ++rt)
            #pragma unroll
            for (int g = 0; g < 4; ++g) {
                float bv = biasv[g];
                acc[rt * 4 + g] = (v8f){bv, bv, bv, bv, bv, bv, bv, bv};
            }

        // pipeline prologue: fragments for the first k-chunk
        v8h a0lo, a0hi, a1lo, a1hi, blo[4], bhi[4];
        {
            const int k0 = (step & 15) * 32;
            a0lo = *(const v8h*)(Acur + aOff0 + k0);
            a0hi = *(const v8h*)(Acur + aOff0 + k0 + 16);
            a1lo = *(const v8h*)(Acur + aOff1 + k0);
            a1hi = *(const v8h*)(Acur + aOff1 + k0 + 16);
            #pragma unroll
            for (int g = 0; g < 4; ++g) {
                const _Float16* bp = WT + (bOffH[g] + k0);
                blo[g] = *(const v8h*)bp;
                bhi[g] = *(const v8h*)(bp + 16);
            }
        }

        #pragma unroll
        for (int kc = 0; kc < 16; ++kc) {
            // issue next chunk's loads before consuming the current chunk
            v8h na0lo, na0hi, na1lo, na1hi, nblo[4], nbhi[4];
            if (kc < 15) {
                const int k0n = ((kc + 1 + step) & 15) * 32;
                na0lo = *(const v8h*)(Acur + aOff0 + k0n);
                na0hi = *(const v8h*)(Acur + aOff0 + k0n + 16);
                na1lo = *(const v8h*)(Acur + aOff1 + k0n);
                na1hi = *(const v8h*)(Acur + aOff1 + k0n + 16);
                #pragma unroll
                for (int g = 0; g < 4; ++g) {
                    const _Float16* bp = WT + (bOffH[g] + k0n);
                    nblo[g] = *(const v8h*)bp;
                    nbhi[g] = *(const v8h*)(bp + 16);
                }
            }

            v16h a0 = mk16(a0lo, a0hi);
            v16h a1 = mk16(a1lo, a1hi);
            #pragma unroll
            for (int g = 0; g < 4; ++g) {
                v16h b = mk16(blo[g], bhi[g]);
                acc[0 * 4 + g] = __builtin_amdgcn_wmma_f32_16x16x32_f16(
                    false, a0, false, b, (short)0, acc[0 * 4 + g], false, false);
                acc[1 * 4 + g] = __builtin_amdgcn_wmma_f32_16x16x32_f16(
                    false, a1, false, b, (short)0, acc[1 * 4 + g], false, false);
            }

            if (kc < 15) {
                a0lo = na0lo; a0hi = na0hi;
                a1lo = na1lo; a1hi = na1hi;
                #pragma unroll
                for (int g = 0; g < 4; ++g) { blo[g] = nblo[g]; bhi[g] = nbhi[g]; }
            }
        }

        // store next x tile (global loads have long since returned)
        if (haveNext) {
            v8h* dst = (v8h*)(Anxt + gRow * KDIM + gChunk * 16);
            dst[0] = xa;
            dst[1] = xb;
        }

        // lane-local gates, state update, max pool; h -> next buffer (f16)
        const int n = wave * 16 + laneN;     // this lane's h column
        #pragma unroll
        for (int rt = 0; rt < 2; ++rt) {
            #pragma unroll
            for (int r = 0; r < 8; ++r) {
                float zi = acc[rt * 4 + 0][r];
                float zf = acc[rt * 4 + 1][r];
                float zg = acc[rt * 4 + 2][r];
                float zo = acc[rt * 4 + 3][r];
                float ig = sigmoidf_(zi);
                float fg = sigmoidf_(zf);
                float gg = fast_tanhf(zg);
                float og = sigmoidf_(zo);
                int idx = rt * 8 + r;
                float c = fg * c_state[idx] + ig * gg;
                c_state[idx] = c;
                float h = og * fast_tanhf(c);
                hmax[idx] = fmaxf(hmax[idx], h);
                int M = rt * 16 + r + laneHi * 8;
                Anxt[M * KDIM + EDIM + n] = (_Float16)h;
            }
        }

        __syncthreads();   // next buffer fully built; cur buffer free
    }

    // write pooled output: out[b][l*512 + dir*256 + n]
    {
        const int n = wave * 16 + laneN;
        #pragma unroll
        for (int rt = 0; rt < 2; ++rt) {
            #pragma unroll
            for (int r = 0; r < 8; ++r) {
                int M = rt * 16 + r + laneHi * 8;
                out[(size_t)(bx * BBLK + M) * (6 * HDIM) + l * 512 + dir * 256 + n] =
                    hmax[rt * 8 + r];
            }
        }
    }
}

extern "C" void kernel_launch(void* const* d_in, const int* in_sizes, int n_in,
                              void* d_out, int out_size, void* d_ws, size_t ws_size,
                              hipStream_t stream) {
    const int*   tokens = (const int*)d_in[0];
    const float* Emb    = (const float*)d_in[1];
    const float* Wf     = (const float*)d_in[2];
    const float* Uf     = (const float*)d_in[3];
    const float* bf     = (const float*)d_in[4];
    const float* Wb     = (const float*)d_in[5];
    const float* Ub     = (const float*)d_in[6];
    const float* bb     = (const float*)d_in[7];
    float* out = (float*)d_out;

    // workspace layout: EmbH f16 (32000*256), then 6 x [1024][512] f16 slabs
    _Float16* EmbH = (_Float16*)d_ws;
    _Float16* WT6  = EmbH + (size_t)VOCAB * EDIM;

    emb_to_h<<<(VOCAB * EDIM) / 256, 256, 0, stream>>>(Emb, EmbH);
    build_wt<<<(6 * GDIM * KDIM) / 256, 256, 0, stream>>>(Wf, Uf, Wb, Ub, WT6);
    bilstm_pool<<<dim3(BATCH / BBLK, 6), 512, 0, stream>>>(tokens, EmbH, WT6,
                                                           bf, bb, out);
}
